// Model_77163382440047
// MI455X (gfx1250) — matrix operand order repaired and checked by reading, not compiled, not run on hardware
//
#include <hip/hip_runtime.h>
#include <hip/hip_bf16.h>
#include <math.h>

// ---- problem constants (from reference) ----
#define NN    1024
#define DDIM  512
#define EE    16384
#define BBAS  4
#define HHEAD 4
#define DH_   256
#define HDH_  (HHEAD*DH_)   // 1024
#define SW_   2
#define WIN_  5
#define CC    1024
#define SSEG  8
#define TT_   8
#define LL_   4
#define NC_   7
#define NEGV  (-1e30f)

typedef __attribute__((ext_vector_type(16))) _Float16 v16h;
typedef __attribute__((ext_vector_type(8)))  _Float16 v8h;
typedef __attribute__((ext_vector_type(8)))  float    v8f;

// =====================================================================
// WMMA GEMM:  C[M,N] (f32) = A[M,K] (f16, row-major) @ Bt[N,K] (f16)
//             + optional bias[N] + optional addend[M,N]; act: 0/1(relu)/2(tanh)
// grid = (N/128, M/64), block = 256 threads (8 wave32s: 4 in M x 2 in N)
// each wave computes a 16x64 slab = 4 accumulators (A-frag reused 4x)
// =====================================================================
__global__ void gemm_wmma_f16(const _Float16* __restrict__ A,
                              const _Float16* __restrict__ Bt,
                              const float* __restrict__ bias,
                              const float* __restrict__ addend,
                              float* __restrict__ C,
                              int M, int N, int K, int act)
{
  const int lane = threadIdx.x & 31;
  const int wave = threadIdx.x >> 5;
  const int m0 = blockIdx.y * 64  + (wave & 3) * 16;
  const int n0 = blockIdx.x * 128 + (wave >> 2) * 64;
  const int row  = lane & 15;
  const int hi16 = lane >> 4;

  // A fragment: lane holds 16 consecutive K of row m (lanes 16-31 -> K+16), matching B
  const _Float16* pA = A + (size_t)(m0 + row) * K + hi16 * 16;
  // B fragment: lane holds 16 consecutive K of column n (lanes 16-31 -> K+16)
  const _Float16* pB = Bt + (size_t)(n0 + row) * K + hi16 * 16;
  const size_t bstep = (size_t)16 * K;   // next 16-column subtile

  v8f acc0 = {}, acc1 = {}, acc2 = {}, acc3 = {};
  for (int k = 0; k < K; k += 32) {
    // prefetch the streaming A row ~512B ahead (global_prefetch_b8)
    if ((k & 63) == 0) __builtin_prefetch(pA + k + 256, 0, 1);
    v8h alo = *(const v8h*)(pA + k);
    v8h ahi = *(const v8h*)(pA + k + 8);
    v16h a  = __builtin_shufflevector(alo, ahi, 0,1,2,3,4,5,6,7,8,9,10,11,12,13,14,15);
    v16h b0 = *(const v16h*)(pB + k);
    v16h b1 = *(const v16h*)(pB + bstep + k);
    v16h b2 = *(const v16h*)(pB + 2 * bstep + k);
    v16h b3 = *(const v16h*)(pB + 3 * bstep + k);
    acc0 = __builtin_amdgcn_wmma_f32_16x16x32_f16(false, a, false, b0, (short)0, acc0, false, false);
    acc1 = __builtin_amdgcn_wmma_f32_16x16x32_f16(false, a, false, b1, (short)0, acc1, false, false);
    acc2 = __builtin_amdgcn_wmma_f32_16x16x32_f16(false, a, false, b2, (short)0, acc2, false, false);
    acc3 = __builtin_amdgcn_wmma_f32_16x16x32_f16(false, a, false, b3, (short)0, acc3, false, false);
  }

  v8f* accs[4] = { &acc0, &acc1, &acc2, &acc3 };
#pragma unroll
  for (int j = 0; j < 4; ++j) {
    const int col = n0 + 16 * j + row;
    const float bv = bias ? bias[col] : 0.f;
    const v8f av = *accs[j];
#pragma unroll
    for (int r = 0; r < 8; ++r) {
      const int rr = m0 + r + 8 * hi16;        // C/D layout: VGPR r -> M = r (+8 for hi lanes)
      const size_t i = (size_t)rr * N + col;
      float v = av[r] + bv + (addend ? addend[i] : 0.f);
      if (act == 1)      v = fmaxf(v, 0.f);
      else if (act == 2) v = tanhf(v);
      C[i] = v;
    }
  }
}

// ---- f32 -> f16 elementwise ----
__global__ void cvt_f16(_Float16* __restrict__ dst, const float* __restrict__ src, size_t n) {
  for (size_t i = (size_t)blockIdx.x * blockDim.x + threadIdx.x; i < n;
       i += (size_t)gridDim.x * blockDim.x)
    dst[i] = (_Float16)src[i];
}

// ---- transpose+convert: src f32 [K,N] -> dst f16 [N,K] ----
__global__ void trans_cvt(_Float16* __restrict__ dst, const float* __restrict__ src, int K, int N) {
  size_t total = (size_t)K * N;
  for (size_t i = (size_t)blockIdx.x * blockDim.x + threadIdx.x; i < total;
       i += (size_t)gridDim.x * blockDim.x) {
    size_t n = i / K, k = i % K;
    dst[i] = (_Float16)src[k * (size_t)N + n];
  }
}

__global__ void fill0(float* __restrict__ p, size_t n) {
  for (size_t i = (size_t)blockIdx.x * blockDim.x + threadIdx.x; i < n;
       i += (size_t)gridDim.x * blockDim.x) p[i] = 0.f;
}

// ---- RGCN edge messages + segment-sum scatter. grid=E, block=512 ----
__global__ void rgcn_msg(const float* __restrict__ xb,   // [B][N][D]
                         const int* __restrict__ src, const int* __restrict__ dst,
                         const int* __restrict__ etype, const float* __restrict__ comp, // [R][B]
                         float* __restrict__ agg)
{
  const int e = blockIdx.x;
  const int d = threadIdx.x;
  const int s = src[e], dn = dst[e], r = etype[e];
  const float c0 = comp[r*BBAS+0], c1 = comp[r*BBAS+1], c2 = comp[r*BBAS+2], c3 = comp[r*BBAS+3];
  const size_t sd = (size_t)s * DDIM + d;
  float m = c0 * xb[sd]
          + c1 * xb[(size_t)1*NN*DDIM + sd]
          + c2 * xb[(size_t)2*NN*DDIM + sd]
          + c3 * xb[(size_t)3*NN*DDIM + sd];
  atomicAdd(&agg[(size_t)dn * DDIM + d], m);
}

// ---- windowed gather: xw16[(n*WIN+w)] = x[clamp(n+w-SW)] ; grid=N*WIN, block=512 ----
__global__ void win_gather(const float* __restrict__ x, _Float16* __restrict__ xw) {
  const int i = blockIdx.x;
  const int n = i / WIN_, w = i % WIN_;
  int rs = n + w - SW_;
  rs = rs < 0 ? 0 : (rs > NN - 1 ? NN - 1 : rs);
  xw[(size_t)i * DDIM + threadIdx.x] = (_Float16)x[(size_t)rs * DDIM + threadIdx.x];
}

// ---- 5x5 windowed attention, one wave per (n,h). block=128 (4 waves) ----
__global__ void attn(const float* __restrict__ q, const float* __restrict__ k,
                     const float* __restrict__ v, _Float16* __restrict__ ao)
{
  const int wid  = ((int)blockIdx.x * blockDim.x + (int)threadIdx.x) >> 5;
  const int lane = threadIdx.x & 31;
  const int n = wid / HHEAD, h = wid % HHEAD;
  const size_t base = (size_t)n * WIN_ * HDH_ + (size_t)h * DH_;

  const int w = lane / 5, u = lane % 5;
  float s = 0.f;
  if (lane < 25) {
    const float4* qp = (const float4*)(q + base + (size_t)w * HDH_);
    const float4* kp = (const float4*)(k + base + (size_t)u * HDH_);
    float acc = 0.f;
    for (int d = 0; d < DH_ / 4; ++d) {
      const float4 qv = qp[d], kv = kp[d];
      acc += qv.x * kv.x + qv.y * kv.y + qv.z * kv.z + qv.w * kv.w;
    }
    s = acc * (1.0f / 16.0f);    // /sqrt(256)
  }
  const int wrow = (lane < 25) ? w : 0;
  float m = -INFINITY;
  for (int j = 0; j < 5; ++j) m = fmaxf(m, __shfl(s, wrow * 5 + j, 32));
  float e = (lane < 25) ? expf(s - m) : 0.f;
  float sum = 0.f;
  for (int j = 0; j < 5; ++j) sum += __shfl(e, wrow * 5 + j, 32);
  float a = (lane < 25) ? e / sum : 0.f;

  for (int ww = 0; ww < WIN_; ++ww) {
    float av[5];
#pragma unroll
    for (int j = 0; j < 5; ++j) av[j] = __shfl(a, ww * 5 + j, 32);
#pragma unroll
    for (int i = 0; i < DH_ / 32; ++i) {
      const int dh = lane + i * 32;
      float o = 0.f;
#pragma unroll
      for (int j = 0; j < 5; ++j) o += av[j] * v[base + (size_t)j * HDH_ + dh];
      ao[base + (size_t)ww * HDH_ + dh] = (_Float16)o;
    }
  }
}

// ---- mean over window: grid=N, block=512 ----
__global__ void win_mean(const float* __restrict__ oW, float* __restrict__ out) {
  const int n = blockIdx.x, d = threadIdx.x;
  float acc = 0.f;
  for (int w = 0; w < WIN_; ++w) acc += oW[((size_t)n * WIN_ + w) * DDIM + d];
  out[(size_t)n * DDIM + d] = acc * (1.f / WIN_);
}

// ---- glove masked-mean pooling -> f16 ; grid=rows, block=512 ----
__global__ void pool_emb(const float* __restrict__ glove, const int* __restrict__ ids,
                         const float* __restrict__ sel, _Float16* __restrict__ out)
{
  const int r = blockIdx.x, d = threadIdx.x;
  float den = 0.f, num = 0.f;
#pragma unroll
  for (int l = 0; l < LL_; ++l) {
    const float sl = sel[(size_t)r * LL_ + l];
    den += sl;
    const int id = ids[(size_t)r * LL_ + l];
    num += sl * glove[(size_t)id * DDIM + d];
  }
  out[(size_t)r * DDIM + d] = (_Float16)(num / fmaxf(den, 1.0f));
}

// ---- symbolic stage 1: alpha over T, src_emb2 ; grid=C*S, block=256 ----
__global__ void sym1(const float* __restrict__ src_emb, const float* __restrict__ dst_emb,
                     const float* __restrict__ wgt, const float* __restrict__ sentics,
                     const float* __restrict__ masks, const int* __restrict__ rels,
                     const float* __restrict__ rvec, const float* __restrict__ src_masks,
                     float* __restrict__ src_emb2)
{
  const int cs = blockIdx.x;
  __shared__ float red[256];
  __shared__ float alpha[TT_];
  __shared__ float wfac[TT_];
  __shared__ float mk[TT_];
  __shared__ int   rl[TT_];
  if (threadIdx.x < TT_) {
    const int t = threadIdx.x;
    const size_t idx = (size_t)cs * TT_ + t;
    mk[t] = masks[idx];
    wfac[t] = wgt[idx] * (1.f + sentics[idx]) * mk[t];
    rl[t] = rels[idx];
  }
  __syncthreads();
  const float* se = src_emb + (size_t)cs * DDIM;
  for (int t = 0; t < TT_; ++t) {
    float acc = 0.f;
    for (int d = threadIdx.x; d < DDIM; d += 256) {
      const float rd = rvec[(size_t)rl[t] * DDIM + d]
                     * dst_emb[((size_t)cs * TT_ + t) * DDIM + d] * wfac[t];
      acc += se[d] * rd;
    }
    red[threadIdx.x] = acc; __syncthreads();
    for (int off = 128; off; off >>= 1) {
      if (threadIdx.x < off) red[threadIdx.x] += red[threadIdx.x + off];
      __syncthreads();
    }
    if (threadIdx.x == 0) alpha[t] = red[0];
    __syncthreads();
  }
  if (threadIdx.x == 0) {
    float m = -INFINITY; float ex[TT_];
    for (int t = 0; t < TT_; ++t) {
      const float sc = mk[t] > 0.f ? alpha[t] : NEGV;
      m = fmaxf(m, sc);
    }
    float sum = 0.f;
    for (int t = 0; t < TT_; ++t) {
      const float sc = mk[t] > 0.f ? alpha[t] : NEGV;
      ex[t] = expf(sc - m); sum += ex[t];
    }
    const float sm = src_masks[cs];
    for (int t = 0; t < TT_; ++t) alpha[t] = ex[t] / sum * sm;
  }
  __syncthreads();
  for (int d = threadIdx.x; d < DDIM; d += 256) {
    float acc = se[d];
    for (int t = 0; t < TT_; ++t) {
      const float rd = rvec[(size_t)rl[t] * DDIM + d]
                     * dst_emb[((size_t)cs * TT_ + t) * DDIM + d] * wfac[t];
      acc += alpha[t] * rd;
    }
    src_emb2[(size_t)cs * DDIM + d] = acc;
  }
}

// ---- symbolic stage 2: att over S, symbolic[c] ; grid=C, block=256 ----
__global__ void sym2(const float* __restrict__ src_emb2, const float* __restrict__ u,
                     const float* __restrict__ masks, const float* __restrict__ src_masks,
                     float* __restrict__ symbolic)
{
  const int c = blockIdx.x;
  __shared__ float red[256];
  __shared__ float att[SSEG];
  for (int s = 0; s < SSEG; ++s) {
    float acc = 0.f;
    for (int d = threadIdx.x; d < DDIM; d += 256)
      acc += src_emb2[((size_t)c * SSEG + s) * DDIM + d] * u[(size_t)c * DDIM + d];
    red[threadIdx.x] = acc; __syncthreads();
    for (int off = 128; off; off >>= 1) {
      if (threadIdx.x < off) red[threadIdx.x] += red[threadIdx.x + off];
      __syncthreads();
    }
    if (threadIdx.x == 0) att[s] = red[0];
    __syncthreads();
  }
  if (threadIdx.x == 0) {
    float m = -INFINITY; float sc[SSEG]; float ex[SSEG];
    for (int s = 0; s < SSEG; ++s) {
      float msum = 0.f;
      for (int t = 0; t < TT_; ++t) msum += masks[((size_t)c * SSEG + s) * TT_ + t];
      sc[s] = msum > 0.f ? att[s] : NEGV;
      m = fmaxf(m, sc[s]);
    }
    float sum = 0.f;
    for (int s = 0; s < SSEG; ++s) { ex[s] = expf(sc[s] - m); sum += ex[s]; }
    for (int s = 0; s < SSEG; ++s) att[s] = ex[s] / sum * src_masks[(size_t)c * SSEG + s];
  }
  __syncthreads();
  for (int d = threadIdx.x; d < DDIM; d += 256) {
    float acc = 0.f;
    for (int s = 0; s < SSEG; ++s)
      acc += att[s] * src_emb2[((size_t)c * SSEG + s) * DDIM + d];
    symbolic[(size_t)c * DDIM + d] = acc;
  }
}

// ---- feat = concat([x, hrgcn, relatt, symbolic]) as f16 ; grid=N, block=256 ----
__global__ void feat_build(const float* __restrict__ x, const float* __restrict__ hrg,
                           const float* __restrict__ rel, const float* __restrict__ sym,
                           _Float16* __restrict__ feat)
{
  const int n = blockIdx.x;
  const size_t o = (size_t)n * 4 * DDIM;
  for (int d = threadIdx.x; d < DDIM; d += 256) {
    feat[o + d]             = (_Float16)x  [(size_t)n * DDIM + d];
    feat[o + DDIM + d]      = (_Float16)hrg[(size_t)n * DDIM + d];
    feat[o + 2 * DDIM + d]  = (_Float16)rel[(size_t)n * DDIM + d];
    feat[o + 3 * DDIM + d]  = (_Float16)sym[(size_t)n * DDIM + d];
  }
}

// ---- head: logits = z @ Wlin + blin, log_softmax ; grid=N, block=256 ----
__global__ void head(const float* __restrict__ z, const float* __restrict__ Wlin,
                     const float* __restrict__ blin, float* __restrict__ out)
{
  const int n = blockIdx.x;
  __shared__ float red[256];
  __shared__ float logits[NC_];
  for (int c = 0; c < NC_; ++c) {
    float acc = 0.f;
    for (int d = threadIdx.x; d < DDIM; d += 256)
      acc += z[(size_t)n * DDIM + d] * Wlin[(size_t)d * NC_ + c];
    red[threadIdx.x] = acc; __syncthreads();
    for (int off = 128; off; off >>= 1) {
      if (threadIdx.x < off) red[threadIdx.x] += red[threadIdx.x + off];
      __syncthreads();
    }
    if (threadIdx.x == 0) logits[c] = red[0] + blin[c];
    __syncthreads();
  }
  if (threadIdx.x == 0) {
    float m = -INFINITY;
    for (int c = 0; c < NC_; ++c) m = fmaxf(m, logits[c]);
    float sum = 0.f;
    for (int c = 0; c < NC_; ++c) sum += expf(logits[c] - m);
    const float lse = m + logf(sum);
    for (int c = 0; c < NC_; ++c) out[(size_t)n * NC_ + c] = logits[c] - lse;
  }
}

// =====================================================================
// Orchestration
// =====================================================================
static inline void launch_gemm(hipStream_t s, const _Float16* A, const _Float16* Bt,
                               const float* bias, const float* add, float* C,
                               int M, int N, int K, int act)
{
  dim3 g(N / 128, M / 64);
  gemm_wmma_f16<<<g, 256, 0, s>>>(A, Bt, bias, add, C, M, N, K, act);
}

extern "C" void kernel_launch(void* const* d_in, const int* in_sizes, int n_in,
                              void* d_out, int out_size, void* d_ws, size_t ws_size,
                              hipStream_t stream)
{
  const float* x      = (const float*)d_in[0];
  const int*   srcE   = (const int*)  d_in[1];
  const int*   dstE   = (const int*)  d_in[2];
  const int*   etype  = (const int*)  d_in[3];
  const float* comp   = (const float*)d_in[4];
  const float* V1     = (const float*)d_in[5];
  const float* V2     = (const float*)d_in[6];
  const float* Wl1    = (const float*)d_in[7];
  const float* b1     = (const float*)d_in[8];
  const float* Wl2    = (const float*)d_in[9];
  const float* b2     = (const float*)d_in[10];
  const float* Wq     = (const float*)d_in[11];
  const float* Wk     = (const float*)d_in[12];
  const float* Wv     = (const float*)d_in[13];
  const float* Wo     = (const float*)d_in[14];
  const float* glove  = (const float*)d_in[15];
  const float* Wc     = (const float*)d_in[16];
  const float* bc     = (const float*)d_in[17];
  const int*   src_ids= (const int*)  d_in[18];
  const int*   dst_ids= (const int*)  d_in[19];
  const float* src_sel= (const float*)d_in[20];
  const float* dst_sel= (const float*)d_in[21];
  const float* wgt    = (const float*)d_in[22];
  const float* sentics= (const float*)d_in[23];
  const float* masks  = (const float*)d_in[24];
  const float* src_m  = (const float*)d_in[25];
  const int*   rels   = (const int*)  d_in[26];
  const float* rvec   = (const float*)d_in[27];
  const float* Wfus   = (const float*)d_in[28];
  const float* bfus   = (const float*)d_in[29];
  const float* Wlin   = (const float*)d_in[30];
  const float* blin   = (const float*)d_in[31];
  float* out = (float*)d_out;

  // ---- workspace carve-up ----
  char* w = (char*)d_ws;
  auto alloc = [&](size_t bytes) -> void* {
    void* p = (void*)w;
    w += (bytes + 255) & ~(size_t)255;
    return p;
  };
  const size_t DD2 = (size_t)DDIM * DDIM;
  _Float16* x16    = (_Float16*)alloc((size_t)NN * DDIM * 2);
  _Float16* h16    = (_Float16*)alloc((size_t)NN * DDIM * 2);
  _Float16* V1t    = (_Float16*)alloc((size_t)BBAS * DD2 * 2);
  _Float16* V2t    = (_Float16*)alloc((size_t)BBAS * DD2 * 2);
  _Float16* Wl1t   = (_Float16*)alloc(DD2 * 2);
  _Float16* Wl2t   = (_Float16*)alloc(DD2 * 2);
  _Float16* Wqt    = (_Float16*)alloc((size_t)DDIM * HDH_ * 2);
  _Float16* Wkt    = (_Float16*)alloc((size_t)DDIM * HDH_ * 2);
  _Float16* Wvt    = (_Float16*)alloc((size_t)DDIM * HDH_ * 2);
  _Float16* Wot    = (_Float16*)alloc((size_t)DDIM * HDH_ * 2);
  _Float16* Wct    = (_Float16*)alloc(DD2 * 2);
  _Float16* Wfust  = (_Float16*)alloc((size_t)4 * DD2 * 2);
  float*    xb     = (float*)alloc((size_t)BBAS * NN * DDIM * 4);
  float*    agg    = (float*)alloc((size_t)NN * DDIM * 4);
  float*    hidden = (float*)alloc((size_t)NN * DDIM * 4);
  float*    hrgcn  = (float*)alloc((size_t)NN * DDIM * 4);
  _Float16* xw16   = (_Float16*)alloc((size_t)NN * WIN_ * DDIM * 2);
  float*    qb     = (float*)alloc((size_t)NN * WIN_ * HDH_ * 4);
  float*    kb     = (float*)alloc((size_t)NN * WIN_ * HDH_ * 4);
  float*    vb     = (float*)alloc((size_t)NN * WIN_ * HDH_ * 4);
  _Float16* ao16   = (_Float16*)alloc((size_t)NN * WIN_ * HDH_ * 2);
  float*    oW     = (float*)alloc((size_t)NN * WIN_ * DDIM * 4);
  float*    relatt = (float*)alloc((size_t)NN * DDIM * 4);
  _Float16* psrc16 = (_Float16*)alloc((size_t)CC * SSEG * DDIM * 2);
  _Float16* pdst16 = (_Float16*)alloc((size_t)CC * SSEG * TT_ * DDIM * 2);
  float*    srcEmb = (float*)alloc((size_t)CC * SSEG * DDIM * 4);
  float*    dstEmb = (float*)alloc((size_t)CC * SSEG * TT_ * DDIM * 4);
  float*    srcE2  = (float*)alloc((size_t)CC * SSEG * DDIM * 4);
  float*    symb   = (float*)alloc((size_t)CC * DDIM * 4);
  _Float16* feat16 = (_Float16*)alloc((size_t)NN * 4 * DDIM * 2);
  float*    zbuf   = (float*)alloc((size_t)NN * DDIM * 4);
  (void)ws_size; (void)in_sizes; (void)n_in; (void)out_size;

  // ---- weight transposes (f32 [K,N] -> f16 [N,K]) ----
  for (int b = 0; b < BBAS; ++b) {
    trans_cvt<<<1024, 256, 0, stream>>>(V1t + b * DD2, V1 + b * DD2, DDIM, DDIM);
    trans_cvt<<<1024, 256, 0, stream>>>(V2t + b * DD2, V2 + b * DD2, DDIM, DDIM);
  }
  trans_cvt<<<1024, 256, 0, stream>>>(Wl1t, Wl1, DDIM, DDIM);
  trans_cvt<<<1024, 256, 0, stream>>>(Wl2t, Wl2, DDIM, DDIM);
  trans_cvt<<<2048, 256, 0, stream>>>(Wqt, Wq, DDIM, HDH_);
  trans_cvt<<<2048, 256, 0, stream>>>(Wkt, Wk, DDIM, HDH_);
  trans_cvt<<<2048, 256, 0, stream>>>(Wvt, Wv, DDIM, HDH_);
  trans_cvt<<<2048, 256, 0, stream>>>(Wot, Wo, HDH_, DDIM);
  trans_cvt<<<1024, 256, 0, stream>>>(Wct, Wc, DDIM, DDIM);
  trans_cvt<<<4096, 256, 0, stream>>>(Wfust, Wfus, 4 * DDIM, DDIM);

  cvt_f16<<<2048, 256, 0, stream>>>(x16, x, (size_t)NN * DDIM);

  // ---- RGCN layer 1 ----
  for (int b = 0; b < BBAS; ++b)
    launch_gemm(stream, x16, V1t + b * DD2, nullptr, nullptr,
                xb + (size_t)b * NN * DDIM, NN, DDIM, DDIM, 0);
  fill0<<<2048, 256, 0, stream>>>(agg, (size_t)NN * DDIM);
  rgcn_msg<<<EE, DDIM, 0, stream>>>(xb, srcE, dstE, etype, comp, agg);
  launch_gemm(stream, x16, Wl1t, b1, agg, hidden, NN, DDIM, DDIM, /*relu*/1);
  cvt_f16<<<2048, 256, 0, stream>>>(h16, hidden, (size_t)NN * DDIM);

  // ---- RGCN layer 2 ----
  for (int b = 0; b < BBAS; ++b)
    launch_gemm(stream, h16, V2t + b * DD2, nullptr, nullptr,
                xb + (size_t)b * NN * DDIM, NN, DDIM, DDIM, 0);
  fill0<<<2048, 256, 0, stream>>>(agg, (size_t)NN * DDIM);
  rgcn_msg<<<EE, DDIM, 0, stream>>>(xb, srcE, dstE, etype, comp, agg);
  launch_gemm(stream, h16, Wl2t, b2, agg, hrgcn, NN, DDIM, DDIM, 0);

  // ---- windowed relational attention ----
  win_gather<<<NN * WIN_, DDIM, 0, stream>>>(x, xw16);
  launch_gemm(stream, xw16, Wqt, nullptr, nullptr, qb, NN * WIN_, HDH_, DDIM, 0);
  launch_gemm(stream, xw16, Wkt, nullptr, nullptr, kb, NN * WIN_, HDH_, DDIM, 0);
  launch_gemm(stream, xw16, Wvt, nullptr, nullptr, vb, NN * WIN_, HDH_, DDIM, 0);
  attn<<<(NN * HHEAD) / 4, 128, 0, stream>>>(qb, kb, vb, ao16);
  launch_gemm(stream, ao16, Wot, nullptr, nullptr, oW, NN * WIN_, DDIM, HDH_, 0);
  win_mean<<<NN, DDIM, 0, stream>>>(oW, relatt);

  // ---- concept embeddings ----
  pool_emb<<<CC * SSEG, DDIM, 0, stream>>>(glove, src_ids, src_sel, psrc16);
  pool_emb<<<CC * SSEG * TT_, DDIM, 0, stream>>>(glove, dst_ids, dst_sel, pdst16);
  launch_gemm(stream, psrc16, Wct, bc, nullptr, srcEmb, CC * SSEG, DDIM, DDIM, 0);
  launch_gemm(stream, pdst16, Wct, bc, nullptr, dstEmb, CC * SSEG * TT_, DDIM, DDIM, 0);

  // ---- symbolic attention ----
  sym1<<<CC * SSEG, 256, 0, stream>>>(srcEmb, dstEmb, wgt, sentics, masks, rels, rvec,
                                      src_m, srcE2);
  sym2<<<CC, 256, 0, stream>>>(srcE2, relatt, masks, src_m, symb);

  // ---- fusion + head ----
  feat_build<<<NN, 256, 0, stream>>>(x, hrgcn, relatt, symb, feat16);
  launch_gemm(stream, feat16, Wfust, bfus, nullptr, zbuf, NN, DDIM, 4 * DDIM, /*tanh*/2);
  head<<<NN, 256, 0, stream>>>(zbuf, Wlin, blin, out);
}
